// MoNet_3968549781763
// MI455X (gfx1250) — matrix operand order repaired and checked by reading, not compiled, not run on hardware
//
#include <hip/hip_runtime.h>
#include <hip/hip_bf16.h>

typedef __attribute__((ext_vector_type(16))) _Float16 v16h;
typedef __attribute__((ext_vector_type(8)))  float    v8f;

#define NN 131072
#define EE 1048576
#define BB 64
#define KK 25
#define GEPS 1e-15f

__device__ __forceinline__ float eluf(float x) { return x > 0.f ? x : (__expf(x) - 1.f); }

// ---------------- precompute gaussian params: gp[k] = {mux, muy, 0.5/(eps+sx^2), 0.5/(eps+sy^2)} ----------------
__global__ void prep_gauss(const float* __restrict__ mu, const float* __restrict__ sg,
                           float* __restrict__ gp) {
  int k = threadIdx.x;
  if (k >= KK) return;
  float sx = sg[2 * k], sy = sg[2 * k + 1];
  gp[4 * k + 0] = mu[2 * k];
  gp[4 * k + 1] = mu[2 * k + 1];
  gp[4 * k + 2] = 0.5f / (GEPS + sx * sx);
  gp[4 * k + 3] = 0.5f / (GEPS + sy * sy);
}

// ---------------- Layer 1 (Cin=1): per-edge VALU kernel ----------------
__global__ void l1_edge(const float* __restrict__ x,
                        const int* __restrict__ ei0, const int* __restrict__ ei1,
                        const float* __restrict__ ea,
                        const float* __restrict__ gp,
                        const float* __restrict__ W1,
                        float* __restrict__ agg, float* __restrict__ cnt) {
  int e = blockIdx.x * blockDim.x + threadIdx.x;
  if (e >= EE) return;
  int s = ei0[e], d = ei1[e];
  if (s == d) return;                       // valid == 0: no msg, no cnt
  float px = ea[2 * e], py = ea[2 * e + 1];
  float xs = x[s];
  float w[32];
#pragma unroll
  for (int c = 0; c < 32; ++c) w[c] = 0.f;
  for (int k = 0; k < KK; ++k) {
    float dx = px - gp[4 * k], dy = py - gp[4 * k + 1];
    float gk = __expf(-(dx * dx * gp[4 * k + 2] + dy * dy * gp[4 * k + 3]));
#pragma unroll
    for (int c = 0; c < 32; ++c) w[c] += gk * W1[k * 32 + c];
  }
  float* row = agg + (long)d * 32;
#pragma unroll
  for (int c = 0; c < 32; ++c) atomicAdd(row + c, xs * w[c]);
  atomicAdd(cnt + d, 1.f);
}

// ---------------- generic node update: act(agg/cnt + hin@root + b) ----------------
__global__ void node_update(const float* __restrict__ agg, const float* __restrict__ cnt,
                            const float* __restrict__ hin, const float* __restrict__ root,
                            const float* __restrict__ bias, float* __restrict__ hout,
                            int nNodes, int Cin, int Cout, int act) {
  int i = blockIdx.x * blockDim.x + threadIdx.x;
  if (i >= nNodes * Cout) return;
  int n = i / Cout, c = i % Cout;
  float r = 0.f;
  for (int ci = 0; ci < Cin; ++ci) r += hin[(long)n * Cin + ci] * root[ci * Cout + c];
  float v = agg[(long)n * Cout + c] / fmaxf(cnt[n], 1.f) + r + bias[c];
  hout[i] = act ? eluf(v) : fmaxf(v, 0.f);
}

// ---------------- pair pooling: h=max of pair, pos=mean of pair ----------------
__global__ void pool_pairs(const float* __restrict__ hin, const float* __restrict__ posin,
                           float* __restrict__ hout, float* __restrict__ posout,
                           int n2, int C) {
  int i = blockIdx.x * blockDim.x + threadIdx.x;
  if (i < n2 * C) {
    int n = i / C, c = i % C;
    hout[i] = fmaxf(hin[(long)(2 * n) * C + c], hin[(long)(2 * n + 1) * C + c]);
  }
  if (i < n2 * 2) {
    int n = i >> 1, j = i & 1;
    posout[i] = 0.5f * (posin[4 * n + j] + posin[4 * n + 2 + j]);
  }
}

// ---------------- global max |pos[dst]-pos[src]| ----------------
__global__ void max_red(const int* __restrict__ ei0, const int* __restrict__ ei1, int shift,
                        const float* __restrict__ posL, unsigned* __restrict__ maxm) {
  float mx = 0.f;
  for (int e = blockIdx.x * blockDim.x + threadIdx.x; e < EE; e += gridDim.x * blockDim.x) {
    int s = ei0[e] >> shift, d = ei1[e] >> shift;
    mx = fmaxf(mx, fabsf(posL[2 * d] - posL[2 * s]));
    mx = fmaxf(mx, fabsf(posL[2 * d + 1] - posL[2 * s + 1]));
  }
  __shared__ float red[256];
  red[threadIdx.x] = mx;
  __syncthreads();
  for (int o = 128; o > 0; o >>= 1) {
    if (threadIdx.x < (unsigned)o) red[threadIdx.x] = fmaxf(red[threadIdx.x], red[threadIdx.x + o]);
    __syncthreads();
  }
  if (threadIdx.x == 0) atomicMax(maxm, __float_as_uint(red[0]));
}

// ---------------- pack Wflat[K*Cin,64] -> f16 B tiles in 32x16 layout, contiguous k ----------------
// tile (kt,nt): lane l holds N=(l&15)+16*nt, K = kt*32 + j + 16*(l>>4), j=0..15 contiguous.
__global__ void pack_w(const float* __restrict__ W, _Float16* __restrict__ out, int KT) {
  int idx = blockIdx.x * blockDim.x + threadIdx.x;
  int total = KT * 4 * 32 * 16;
  if (idx >= total) return;
  int j    = idx & 15;
  int lane = (idx >> 4) & 31;
  int nt   = (idx >> 9) & 3;
  int kt   = idx >> 11;
  int kk = kt * 32 + j + ((lane >> 4) << 4);
  int co = nt * 16 + (lane & 15);
  out[idx] = (_Float16)W[(long)kk * 64 + co];
}

// ---------------- edge GEMM: msg[16e x 64] = f[16e x K*Cin] @ Wflat, WMMA f16 ----------------
template <int CIN>
__launch_bounds__(256)
__global__ void gmm_gemm(const int* __restrict__ ei0, const int* __restrict__ ei1, int shift,
                         const float* __restrict__ xsrc, const float* __restrict__ posL,
                         const unsigned* __restrict__ maxm,
                         const float* __restrict__ gp,
                         const _Float16* __restrict__ wpack,
                         float* __restrict__ agg, float* __restrict__ cnt, int nTiles) {
  constexpr int KT_LDS = 25;              // B tiles staged in LDS: 25*4*32*32B = 100 KB
  __shared__ v16h  ldsW[KT_LDS * 4 * 32];
  __shared__ float ldsG[8 * 16 * KK];     // per-wave gaussian weights: 12.8 KB

  // cooperative stage of the first KT_LDS k-chunks of packed W into LDS
  {
    const uint4* s4 = (const uint4*)wpack;
    uint4* d4 = (uint4*)ldsW;
    const int n4 = KT_LDS * 4 * 32 * 2;   // two uint4 per v16h
    for (int i = threadIdx.x; i < n4; i += 256) d4[i] = s4[i];
  }
  __syncthreads();

  const v16h* wglob = (const v16h*)wpack;
  const int tid  = threadIdx.x;
  const int lane = tid & 31;
  const int wave = tid >> 5;
  const int half = lane >> 4;
  const int m    = lane & 15;
  float* gRow = &ldsG[wave * 16 * KK];

  float inv2m;
  { float mv = __uint_as_float(maxm[0]); inv2m = 0.5f / fmaxf(mv, 1e-12f); }

  for (int t = blockIdx.x * 8 + wave; t < nTiles; t += gridDim.x * 8) {
    const int eb = t << 4;
    const int e  = eb + m;
    const int s  = ei0[e] >> shift;

    // --- lanes 0-15 compute gaussian weights for their edge row into LDS ---
    if (half == 0) {
      const int d = ei1[e] >> shift;
      float px = (posL[2 * d]     - posL[2 * s])     * inv2m + 0.5f;
      float py = (posL[2 * d + 1] - posL[2 * s + 1]) * inv2m + 0.5f;
#pragma unroll
      for (int k = 0; k < KK; ++k) {
        float dx = px - gp[4 * k], dy = py - gp[4 * k + 1];
        gRow[m * KK + k] = __expf(-(dx * dx * gp[4 * k + 2] + dy * dy * gp[4 * k + 3]));
      }
    }

    // --- gather x_src for this lane's fixed A-columns (contiguous k = 16*half + j) ---
    const float* xr = xsrc + (long)s * CIN;
    float xlo[16], xhi[16];
#pragma unroll
    for (int j = 0; j < 16; ++j) {
      int c = j + (half << 4);
      xlo[j] = xr[c];
      if (CIN == 64) xhi[j] = xr[32 + c];
    }

    // --- this lane's C-rows (M = r + 8*half): dst + valid mask ---
    int dr[8]; float vf[8];
#pragma unroll
    for (int r = 0; r < 8; ++r) {
      int er = eb + r + (half << 3);
      int sr = ei0[er] >> shift, dd = ei1[er] >> shift;
      dr[r] = dd;
      vf[r] = (sr != dd) ? 1.f : 0.f;
    }

    v8f a0 = {}, a1 = {}, a2 = {}, a3 = {};
#pragma unroll 1
    for (int kk = 0; kk < KK; ++kk) {        // rolled: low register pressure
      const float gk = gRow[m * KK + kk];
      {
        v16h av;
#pragma unroll
        for (int j = 0; j < 16; ++j) av[j] = (_Float16)(gk * xlo[j]);
        const int kt = (CIN == 64) ? 2 * kk : kk;
        v16h w0, w1, w2, w3;
        if (kt < KT_LDS) {
          const v16h* p = &ldsW[kt * 128 + lane];
          w0 = p[0]; w1 = p[32]; w2 = p[64]; w3 = p[96];
        } else {
          const v16h* p = wglob + (long)kt * 128 + lane;
          w0 = p[0]; w1 = p[32]; w2 = p[64]; w3 = p[96];
        }
        a0 = __builtin_amdgcn_wmma_f32_16x16x32_f16(false, av, false, w0, (short)0, a0, false, false);
        a1 = __builtin_amdgcn_wmma_f32_16x16x32_f16(false, av, false, w1, (short)0, a1, false, false);
        a2 = __builtin_amdgcn_wmma_f32_16x16x32_f16(false, av, false, w2, (short)0, a2, false, false);
        a3 = __builtin_amdgcn_wmma_f32_16x16x32_f16(false, av, false, w3, (short)0, a3, false, false);
      }
      if (CIN == 64) {                       // second 32-wide chunk of this k
        v16h av;
#pragma unroll
        for (int j = 0; j < 16; ++j) av[j] = (_Float16)(gk * xhi[j]);
        const int kt = 2 * kk + 1;
        v16h w0, w1, w2, w3;
        if (kt < KT_LDS) {
          const v16h* p = &ldsW[kt * 128 + lane];
          w0 = p[0]; w1 = p[32]; w2 = p[64]; w3 = p[96];
        } else {
          const v16h* p = wglob + (long)kt * 128 + lane;
          w0 = p[0]; w1 = p[32]; w2 = p[64]; w3 = p[96];
        }
        a0 = __builtin_amdgcn_wmma_f32_16x16x32_f16(false, av, false, w0, (short)0, a0, false, false);
        a1 = __builtin_amdgcn_wmma_f32_16x16x32_f16(false, av, false, w1, (short)0, a1, false, false);
        a2 = __builtin_amdgcn_wmma_f32_16x16x32_f16(false, av, false, w2, (short)0, a2, false, false);
        a3 = __builtin_amdgcn_wmma_f32_16x16x32_f16(false, av, false, w3, (short)0, a3, false, false);
      }
    }

    // --- masked scatter-add into agg (mean aggregation numerator) ---
#pragma unroll
    for (int r = 0; r < 8; ++r) {
      float* row = agg + (long)dr[r] * 64;
      atomicAdd(row + m,      a0[r] * vf[r]);
      atomicAdd(row + 16 + m, a1[r] * vf[r]);
      atomicAdd(row + 32 + m, a2[r] * vf[r]);
      atomicAdd(row + 48 + m, a3[r] * vf[r]);
    }
    if (m == 0) {              // lanes 0,16 cover edges 0-7 / 8-15
#pragma unroll
      for (int r = 0; r < 8; ++r) atomicAdd(cnt + dr[r], vf[r]);
    }
  }
}

// ---------------- head: contiguous batch mean-pool + MLP + log_softmax ----------------
__global__ void head(const float* __restrict__ h3,
                     const float* __restrict__ fw1, const float* __restrict__ fb1,
                     const float* __restrict__ fw2, const float* __restrict__ fb2,
                     float* __restrict__ out) {
  __shared__ float pooled[64], z1[128], z2[10], redv[2];
  int b = blockIdx.x, t = threadIdx.x;
  if (t < 64) {
    float s = 0.f;
    const float* base = h3 + (long)b * 512 * 64 + t;
    for (int i = 0; i < 512; ++i) s += base[(long)i * 64];
    pooled[t] = s * (1.f / 512.f);
  }
  __syncthreads();
  {
    float s = fb1[t];
    for (int c = 0; c < 64; ++c) s += pooled[c] * fw1[c * 128 + t];
    z1[t] = eluf(s);
  }
  __syncthreads();
  if (t < 10) {
    float s = fb2[t];
    for (int j = 0; j < 128; ++j) s += z1[j] * fw2[j * 10 + t];
    z2[t] = s;
  }
  __syncthreads();
  if (t == 0) {
    float mx = z2[0];
    for (int j = 1; j < 10; ++j) mx = fmaxf(mx, z2[j]);
    float se = 0.f;
    for (int j = 0; j < 10; ++j) se += __expf(z2[j] - mx);
    redv[0] = mx; redv[1] = __logf(se);
  }
  __syncthreads();
  if (t < 10) out[b * 10 + t] = z2[t] - redv[0] - redv[1];
}

extern "C" void kernel_launch(void* const* d_in, const int* in_sizes, int n_in,
                              void* d_out, int out_size, void* d_ws, size_t ws_size,
                              hipStream_t stream) {
  const float* x    = (const float*)d_in[0];
  const int*   ei0  = (const int*)d_in[1];
  const int*   ei1  = ei0 + EE;
  const float* ea   = (const float*)d_in[2];
  const float* pos  = (const float*)d_in[3];
  const float* mu1  = (const float*)d_in[5];
  const float* sg1  = (const float*)d_in[6];
  const float* W1   = (const float*)d_in[7];
  const float* rt1  = (const float*)d_in[8];
  const float* b1   = (const float*)d_in[9];
  const float* mu2  = (const float*)d_in[10];
  const float* sg2  = (const float*)d_in[11];
  const float* W2   = (const float*)d_in[12];
  const float* rt2  = (const float*)d_in[13];
  const float* b2   = (const float*)d_in[14];
  const float* mu3  = (const float*)d_in[15];
  const float* sg3  = (const float*)d_in[16];
  const float* W3   = (const float*)d_in[17];
  const float* rt3  = (const float*)d_in[18];
  const float* b3   = (const float*)d_in[19];
  const float* fw1  = (const float*)d_in[20];
  const float* fb1  = (const float*)d_in[21];
  const float* fw2  = (const float*)d_in[22];
  const float* fb2  = (const float*)d_in[23];
  float* out = (float*)d_out;

  const int N2 = NN / 2, N3 = NN / 4;

  // workspace carve (256B aligned)
  char* wsp = (char*)d_ws;
  auto alloc = [&](size_t bytes) { void* p = wsp; wsp += (bytes + 255) & ~(size_t)255; return p; };
  float*    agg  = (float*)alloc((size_t)NN * 32 * 4);          // reused: N*32 == N2*64
  float*    cnt  = (float*)alloc((size_t)NN * 4);
  float*    h1   = (float*)alloc((size_t)NN * 32 * 4);
  float*    h1p  = (float*)alloc((size_t)N2 * 32 * 4);
  float*    pos2 = (float*)alloc((size_t)N2 * 2 * 4);
  float*    h2   = (float*)alloc((size_t)N2 * 64 * 4);
  float*    h2p  = (float*)alloc((size_t)N3 * 64 * 4);
  float*    pos3 = (float*)alloc((size_t)N3 * 2 * 4);
  float*    h3   = (float*)alloc((size_t)N3 * 64 * 4);
  unsigned* mxm  = (unsigned*)alloc(8);
  _Float16* wp2  = (_Float16*)alloc((size_t)25 * 4 * 32 * 16 * 2);
  _Float16* wp3  = (_Float16*)alloc((size_t)50 * 4 * 32 * 16 * 2);
  float*    gp1  = (float*)alloc((size_t)KK * 4 * 4);
  float*    gp2  = (float*)alloc((size_t)KK * 4 * 4);
  float*    gp3  = (float*)alloc((size_t)KK * 4 * 4);

  const int nTiles = EE / 16;

  // gaussian param prep (divides hoisted out of all hot loops)
  prep_gauss<<<1, 32, 0, stream>>>(mu1, sg1, gp1);
  prep_gauss<<<1, 32, 0, stream>>>(mu2, sg2, gp2);
  prep_gauss<<<1, 32, 0, stream>>>(mu3, sg3, gp3);

  // ---- layer 1 ----
  hipMemsetAsync(agg, 0, (size_t)NN * 32 * 4, stream);
  hipMemsetAsync(cnt, 0, (size_t)NN * 4, stream);
  l1_edge<<<EE / 256, 256, 0, stream>>>(x, ei0, ei1, ea, gp1, W1, agg, cnt);
  node_update<<<(NN * 32) / 256, 256, 0, stream>>>(agg, cnt, x, rt1, b1, h1, NN, 1, 32, 1);
  pool_pairs<<<(N2 * 32) / 256, 256, 0, stream>>>(h1, pos, h1p, pos2, N2, 32);

  // ---- layer 2 ----
  hipMemsetAsync(mxm, 0, 8, stream);
  max_red<<<1024, 256, 0, stream>>>(ei0, ei1, 1, pos2, mxm);
  pack_w<<<(25 * 4 * 32 * 16) / 256, 256, 0, stream>>>(W2, wp2, 25);
  hipMemsetAsync(agg, 0, (size_t)N2 * 64 * 4, stream);
  hipMemsetAsync(cnt, 0, (size_t)N2 * 4, stream);
  gmm_gemm<32><<<1024, 256, 0, stream>>>(ei0, ei1, 1, h1p, pos2, mxm, gp2, wp2, agg, cnt, nTiles);
  node_update<<<(N2 * 64) / 256, 256, 0, stream>>>(agg, cnt, h1p, rt2, b2, h2, N2, 32, 64, 0);
  pool_pairs<<<(N3 * 64) / 256, 256, 0, stream>>>(h2, pos2, h2p, pos3, N3, 64);

  // ---- layer 3 ----
  hipMemsetAsync(mxm, 0, 8, stream);
  max_red<<<1024, 256, 0, stream>>>(ei0, ei1, 2, pos3, mxm);
  pack_w<<<(50 * 4 * 32 * 16) / 256, 256, 0, stream>>>(W3, wp3, 50);
  hipMemsetAsync(agg, 0, (size_t)N3 * 64 * 4, stream);
  hipMemsetAsync(cnt, 0, (size_t)N3 * 4, stream);
  gmm_gemm<64><<<1024, 256, 0, stream>>>(ei0, ei1, 2, h2p, pos3, mxm, gp3, wp3, agg, cnt, nTiles);
  node_update<<<(N3 * 64) / 256, 256, 0, stream>>>(agg, cnt, h2p, rt3, b3, h3, N3, 64, 64, 1);

  // ---- head ----
  head<<<BB, 128, 0, stream>>>(h3, fw1, fb1, fw2, fb2, out);
}
